// ForwardPolicy_16947940950104
// MI455X (gfx1250) — matrix operand order repaired and checked by reading, not compiled, not run on hardware
//
#include <hip/hip_runtime.h>
#include <hip/hip_bf16.h>
#include <math.h>

#define N_NODESC 2000
#define N_EDGESC 6000
#define E2C      8000      // edges + N self-loops
#define F_INC    64
#define C1C      256       // HEADS1 * HID
#define HIDC     64
#define NAC      6001      // NUM_ACTIONS
#define NAPC     6016      // NA padded to multiple of 64

typedef __attribute__((ext_vector_type(16))) __bf16 v16bf;
typedef __attribute__((ext_vector_type(8)))  float  v8f;
typedef __attribute__((ext_vector_type(4)))  unsigned int uint4v;

// ---------- helpers ----------
__device__ __forceinline__ unsigned short f2bf(float f) {
  unsigned int u = __float_as_uint(f);
  unsigned int r = (u + 0x7FFFu + ((u >> 16) & 1u)) >> 16;   // RNE
  return (unsigned short)r;
}

__device__ __forceinline__ void atomicMaxF(float* addr, float val) {
  unsigned int* a = (unsigned int*)addr;
  unsigned int old = __float_as_uint(*addr);
  while (__uint_as_float(old) < val) {
    unsigned int assumed = old;
    old = atomicCAS(a, assumed, __float_as_uint(val));
    if (old == assumed) break;
  }
}

__device__ __forceinline__ void edge_decode(int e, const long long* ei,
                                            const float* eattr, const float* lattr,
                                            int& s, int& d, float& ea) {
  if (e < N_EDGESC) {
    s = (int)ei[e];
    d = (int)ei[N_EDGESC + e];
    ea = eattr[e];
  } else {
    s = d = e - N_EDGESC;
    ea = lattr[s];
  }
}

// ---------- utility kernels ----------
__global__ void fill_kernel(float* p, float v, int n) {
  int i = blockIdx.x * 256 + threadIdx.x;
  if (i < n) p[i] = v;
}

__global__ void cvt_bf16_kernel(const float* __restrict__ in, unsigned short* __restrict__ out,
                                int nvalid, int ntotal) {
  int i = blockIdx.x * 256 + threadIdx.x;
  if (i < ntotal) out[i] = (i < nvalid) ? f2bf(in[i]) : (unsigned short)0;
}

// self-loop attr: segment mean of edge_attr by dst
__global__ void loop_accum_kernel(const long long* __restrict__ ei, const float* __restrict__ eattr,
                                  float* __restrict__ cnt, float* __restrict__ asum) {
  int e = blockIdx.x * 256 + threadIdx.x;
  if (e < N_EDGESC) {
    int d = (int)ei[N_EDGESC + e];
    atomicAdd(&cnt[d], 1.0f);
    atomicAdd(&asum[d], eattr[e]);
  }
}

__global__ void loop_fin_kernel(const float* __restrict__ cnt, const float* __restrict__ asum,
                                float* __restrict__ lattr) {
  int i = blockIdx.x * 256 + threadIdx.x;
  if (i < N_NODESC) lattr[i] = asum[i] / fmaxf(cnt[i], 1.0f);
}

// ---------- WMMA GEMM: C[M,NPAD] = A[M,K](bf16) * B[NPAD,K](bf16)^T + bias ----------
// One wave computes a 16x64 tile (4 accumulators) via v_wmma_f32_16x16x32_bf16.
// K/NPAD/NVALID are compile-time: fully unrolled WMMA stream, all fragment loads
// are aligned b128 with constant immediate offsets off precomputed base pointers,
// explicitly double-buffered so loads of step t+1 overlap the 4 WMMAs of step t.
template<int K, int NPAD, int NVALID>
__global__ __launch_bounds__(128)
void wmma_gemm_kernel(const unsigned short* __restrict__ A,   // M x K row-major bf16
                      const unsigned short* __restrict__ B,   // NPAD x K row-major bf16 (zero-padded)
                      const float* __restrict__ bias,         // length >= NVALID
                      float* __restrict__ C,                  // M x NPAD row-major f32
                      int M) {
  constexpr int KSTEPS = K / 32;
  const int wave = threadIdx.x >> 5;
  const int lane = threadIdx.x & 31;
  constexpr int tilesN4 = NPAD >> 6;         // 64-column groups
  const int tilesM = M >> 4;
  int tile = blockIdx.x * 4 + wave;
  if (tile >= tilesM * tilesN4) return;      // wave-uniform
  int tm  = tile / tilesN4;
  int tn4 = tile - tm * tilesN4;

  const int hk  = lane >> 4;                 // wave half
  const int l15 = lane & 15;
  const uint4v* A4 = (const uint4v*)A;
  const uint4v* B4 = (const uint4v*)B;
  // Base pointers in uint4 (16B) units; k-step advance = +4 uint4s (constant offsets).
  // A frag (16x32 bf16): ap[ks*4], ap[ks*4+1]
  const uint4v* ap = A4 + (((tm * 16 + l15) * K) >> 3) + hk * 2;
  // B frag (32x16 bf16): bp[j][ks*4], bp[j][ks*4+1]
  const uint4v* bp0 = B4 + (((tn4 * 64 +  0 + l15) * K) >> 3) + hk * 2;
  const uint4v* bp1 = B4 + (((tn4 * 64 + 16 + l15) * K) >> 3) + hk * 2;
  const uint4v* bp2 = B4 + (((tn4 * 64 + 32 + l15) * K) >> 3) + hk * 2;
  const uint4v* bp3 = B4 + (((tn4 * 64 + 48 + l15) * K) >> 3) + hk * 2;

  union Frag { v16bf v; uint4v q[2]; };
  v8f acc[4] = {{}, {}, {}, {}};

  Frag a_cur, b_cur[4];
  a_cur.q[0] = ap[0];   a_cur.q[1] = ap[1];
  b_cur[0].q[0] = bp0[0]; b_cur[0].q[1] = bp0[1];
  b_cur[1].q[0] = bp1[0]; b_cur[1].q[1] = bp1[1];
  b_cur[2].q[0] = bp2[0]; b_cur[2].q[1] = bp2[1];
  b_cur[3].q[0] = bp3[0]; b_cur[3].q[1] = bp3[1];

#pragma unroll
  for (int ks = 0; ks < KSTEPS; ++ks) {
    Frag a_nxt, b_nxt[4];
    if (ks + 1 < KSTEPS) {                   // compile-time condition
      const int o = (ks + 1) * 4;
      a_nxt.q[0] = ap[o];     a_nxt.q[1] = ap[o + 1];
      b_nxt[0].q[0] = bp0[o]; b_nxt[0].q[1] = bp0[o + 1];
      b_nxt[1].q[0] = bp1[o]; b_nxt[1].q[1] = bp1[o + 1];
      b_nxt[2].q[0] = bp2[o]; b_nxt[2].q[1] = bp2[o + 1];
      b_nxt[3].q[0] = bp3[o]; b_nxt[3].q[1] = bp3[o + 1];
    }
#pragma unroll
    for (int j = 0; j < 4; ++j)
      acc[j] = __builtin_amdgcn_wmma_f32_16x16x32_bf16(false, a_cur.v, false, b_cur[j].v,
                                                       (short)0, acc[j], false, false);
    if (ks + 1 < KSTEPS) {
      a_cur = a_nxt;
#pragma unroll
      for (int j = 0; j < 4; ++j) b_cur[j] = b_nxt[j];
    }
  }

  // D layout: N = lane&15 ; rows tm*16 + half*8 + r
  const int mbase = tm * 16 + hk * 8;
#pragma unroll
  for (int j = 0; j < 4; ++j) {
    int n = tn4 * 64 + j * 16 + l15;
    float bv = (n < NVALID) ? bias[n] : 0.0f;
    float* cp = C + mbase * NPAD + n;
#pragma unroll
    for (int r = 0; r < 8; ++r)
      cp[r * NPAD] = acc[j][r] + bv;
  }
}

// ---------- Layer 1 attention ----------
__global__ void logits1_kernel(const long long* __restrict__ ei, const float* __restrict__ eattr,
                               const float* __restrict__ lattr,
                               const float* __restrict__ xl1, const float* __restrict__ xr1,
                               const float* __restrict__ We1, const float* __restrict__ att1,
                               float* __restrict__ log1, float* __restrict__ m1) {
  int t = blockIdx.x * 256 + threadIdx.x;
  if (t >= E2C * 4) return;
  int e = t >> 2, h = t & 3;
  int s, d; float ea;
  edge_decode(e, ei, eattr, lattr, s, d, ea);
  const float* xls = xl1 + s * C1C + h * HIDC;
  const float* xrd = xr1 + d * C1C + h * HIDC;
  const float* we  = We1 + h * HIDC;
  const float* at  = att1 + h * HIDC;
  float acc = 0.0f;
#pragma unroll 8
  for (int c = 0; c < HIDC; ++c) {
    float g = xls[c] + xrd[c] + we[c] * ea;
    g = (g > 0.0f) ? g : 0.2f * g;            // leaky_relu(0.2)
    acc += at[c] * g;
  }
  log1[t] = acc;
  atomicMaxF(&m1[d * 4 + h], acc);
}

__global__ void exp1_kernel(const long long* __restrict__ ei, const float* __restrict__ eattr,
                            const float* __restrict__ lattr,
                            float* __restrict__ log1, const float* __restrict__ m1,
                            float* __restrict__ den1) {
  int t = blockIdx.x * 256 + threadIdx.x;
  if (t >= E2C * 4) return;
  int e = t >> 2, h = t & 3;
  int s, d; float ea;
  edge_decode(e, ei, eattr, lattr, s, d, ea);
  float w = __expf(log1[t] - m1[d * 4 + h]);
  log1[t] = w;
  atomicAdd(&den1[d * 4 + h], w);
}

__global__ void scatter1_kernel(const long long* __restrict__ ei, const float* __restrict__ eattr,
                                const float* __restrict__ lattr,
                                const float* __restrict__ log1, const float* __restrict__ den1,
                                const float* __restrict__ xl1, float* __restrict__ agg1) {
  int e = blockIdx.x;
  int c = threadIdx.x;                        // 0..255
  int h = c >> 6;
  int s, d; float ea;
  edge_decode(e, ei, eattr, lattr, s, d, ea);
  float a = log1[e * 4 + h] / (den1[d * 4 + h] + 1e-16f);
  atomicAdd(&agg1[d * C1C + c], a * xl1[s * C1C + c]);
}

__global__ void h1_fin_kernel(const float* __restrict__ agg1, const float* __restrict__ b1,
                              unsigned short* __restrict__ h1bf) {
  int i = blockIdx.x * 256 + threadIdx.x;
  if (i < N_NODESC * C1C) {
    float v = agg1[i] + b1[i & (C1C - 1)];
    v = (v > 0.0f) ? v : 0.0f;                // relu
    h1bf[i] = f2bf(v);
  }
}

// ---------- Layer 2 attention (1 head, 6001 channels) ----------
__global__ void logits2_kernel(const long long* __restrict__ ei, const float* __restrict__ eattr,
                               const float* __restrict__ lattr,
                               const float* __restrict__ xl2, const float* __restrict__ xr2,
                               const float* __restrict__ We2, const float* __restrict__ att2,
                               float* __restrict__ log2, float* __restrict__ m2) {
  __shared__ float red[256];
  int e = blockIdx.x;
  int tid = threadIdx.x;
  int s, d; float ea;
  edge_decode(e, ei, eattr, lattr, s, d, ea);
  const float* xls = xl2 + s * NAPC;
  const float* xrd = xr2 + d * NAPC;
  float acc = 0.0f;
  for (int c = tid; c < NAC; c += 256) {
    float g = xls[c] + xrd[c] + We2[c] * ea;
    g = (g > 0.0f) ? g : 0.2f * g;
    acc += att2[c] * g;
  }
  red[tid] = acc;
  __syncthreads();
  for (int st = 128; st > 0; st >>= 1) {
    if (tid < st) red[tid] += red[tid + st];
    __syncthreads();
  }
  if (tid == 0) {
    log2[e] = red[0];
    atomicMaxF(&m2[d], red[0]);
  }
}

__global__ void exp2_kernel(const long long* __restrict__ ei, const float* __restrict__ eattr,
                            const float* __restrict__ lattr,
                            float* __restrict__ log2, const float* __restrict__ m2,
                            float* __restrict__ den2) {
  int e = blockIdx.x * 256 + threadIdx.x;
  if (e >= E2C) return;
  int s, d; float ea;
  edge_decode(e, ei, eattr, lattr, s, d, ea);
  float w = __expf(log2[e] - m2[d]);
  log2[e] = w;
  atomicAdd(&den2[d], w);
}

__global__ void scatter2_kernel(const long long* __restrict__ ei, const float* __restrict__ eattr,
                                const float* __restrict__ lattr,
                                const float* __restrict__ log2, const float* __restrict__ den2,
                                const float* __restrict__ xl2, float* __restrict__ agg2) {
  int e = blockIdx.x;
  int c = blockIdx.y * 256 + threadIdx.x;
  if (c >= NAC) return;
  int s, d; float ea;
  edge_decode(e, ei, eattr, lattr, s, d, ea);
  float a = log2[e] / (den2[d] + 1e-16f);
  atomicAdd(&agg2[d * NAPC + c], a * xl2[s * NAPC + c]);
}

// mean pool over nodes (coalesced across channel threads)
__global__ void pool_kernel(const float* __restrict__ agg2, const float* __restrict__ b2,
                            float* __restrict__ pooled) {
  int c = blockIdx.x * 256 + threadIdx.x;
  if (c >= NAC) return;
  float bv = b2[c];
  float s = 0.0f;
  for (int n = 0; n < N_NODESC; ++n) {
    float v = agg2[n * NAPC + c] + bv;
    s += (v > 0.0f) ? v : 0.0f;               // relu
  }
  pooled[c] = s * (1.0f / (float)N_NODESC);
}

__global__ void softmax_out_kernel(const float* __restrict__ pooled,
                                   const float* __restrict__ alpha,
                                   float* __restrict__ out) {
  __shared__ float red[1024];
  int tid = threadIdx.x;
  float mx = -1e30f;
  for (int c = tid; c < NAC; c += 1024) mx = fmaxf(mx, pooled[c]);
  red[tid] = mx;
  __syncthreads();
  for (int st = 512; st > 0; st >>= 1) {
    if (tid < st) red[tid] = fmaxf(red[tid], red[tid + st]);
    __syncthreads();
  }
  float m = red[0];
  __syncthreads();
  float sm = 0.0f;
  for (int c = tid; c < NAC; c += 1024) sm += __expf(pooled[c] - m);
  red[tid] = sm;
  __syncthreads();
  for (int st = 512; st > 0; st >>= 1) {
    if (tid < st) red[tid] += red[tid + st];
    __syncthreads();
  }
  float denom = red[0];
  for (int c = tid; c < NAC; c += 1024) out[c] = __expf(pooled[c] - m) / denom;
  if (tid == 0) out[NAC] = 1.0f / (1.0f + __expf(-alpha[0]));
}

// ---------- host-side launch ----------
extern "C" void kernel_launch(void* const* d_in, const int* in_sizes, int n_in,
                              void* d_out, int out_size, void* d_ws, size_t ws_size,
                              hipStream_t stream) {
  (void)in_sizes; (void)n_in; (void)out_size; (void)ws_size;

  const float*     x     = (const float*)d_in[0];
  const long long* ei    = (const long long*)d_in[1];
  const float*     eattr = (const float*)d_in[2];
  const float*     alpha = (const float*)d_in[3];
  const float* Wl1 = (const float*)d_in[4];  const float* bl1 = (const float*)d_in[5];
  const float* Wr1 = (const float*)d_in[6];  const float* br1 = (const float*)d_in[7];
  const float* We1 = (const float*)d_in[8];  const float* att1 = (const float*)d_in[9];
  const float* b1  = (const float*)d_in[10];
  const float* Wl2 = (const float*)d_in[11]; const float* bl2 = (const float*)d_in[12];
  const float* Wr2 = (const float*)d_in[13]; const float* br2 = (const float*)d_in[14];
  const float* We2 = (const float*)d_in[15]; const float* att2 = (const float*)d_in[16];
  const float* b2  = (const float*)d_in[17];
  float* out = (float*)d_out;

  // workspace layout (256-byte aligned slices)
  size_t off = 0;
  auto take = [&](size_t bytes) -> void* {
    void* p = (char*)d_ws + off;
    off += (bytes + 255) & ~(size_t)255;
    return p;
  };
  float* cnt   = (float*)take(N_NODESC * 4);
  float* asum  = (float*)take(N_NODESC * 4);
  float* lattr = (float*)take(N_NODESC * 4);
  unsigned short* xbf  = (unsigned short*)take((size_t)N_NODESC * F_INC * 2);
  unsigned short* w1l  = (unsigned short*)take((size_t)C1C * F_INC * 2);
  unsigned short* w1r  = (unsigned short*)take((size_t)C1C * F_INC * 2);
  float* xl1  = (float*)take((size_t)N_NODESC * C1C * 4);
  float* xr1  = (float*)take((size_t)N_NODESC * C1C * 4);
  float* log1 = (float*)take((size_t)E2C * 4 * 4);
  float* m1   = (float*)take((size_t)N_NODESC * 4 * 4);
  float* den1 = (float*)take((size_t)N_NODESC * 4 * 4);
  float* agg1 = (float*)take((size_t)N_NODESC * C1C * 4);
  unsigned short* h1bf = (unsigned short*)take((size_t)N_NODESC * C1C * 2);
  unsigned short* w2l  = (unsigned short*)take((size_t)NAPC * C1C * 2);
  unsigned short* w2r  = (unsigned short*)take((size_t)NAPC * C1C * 2);
  float* xl2  = (float*)take((size_t)N_NODESC * NAPC * 4);
  float* xr2  = (float*)take((size_t)N_NODESC * NAPC * 4);
  float* log2 = (float*)take((size_t)E2C * 4);
  float* m2   = (float*)take((size_t)N_NODESC * 4);
  float* den2 = (float*)take((size_t)N_NODESC * 4);
  float* agg2 = (float*)take((size_t)N_NODESC * NAPC * 4);
  float* pooled = (float*)take((size_t)NAPC * 4);

  auto gridOf = [](long long n) { return dim3((unsigned)((n + 255) / 256)); };

  // init accumulators (workspace is poisoned, never re-zeroed by harness)
  fill_kernel<<<gridOf(N_NODESC), 256, 0, stream>>>(cnt, 0.0f, N_NODESC);
  fill_kernel<<<gridOf(N_NODESC), 256, 0, stream>>>(asum, 0.0f, N_NODESC);
  fill_kernel<<<gridOf(N_NODESC * 4), 256, 0, stream>>>(m1, -1e30f, N_NODESC * 4);
  fill_kernel<<<gridOf(N_NODESC * 4), 256, 0, stream>>>(den1, 0.0f, N_NODESC * 4);
  fill_kernel<<<gridOf((long long)N_NODESC * C1C), 256, 0, stream>>>(agg1, 0.0f, N_NODESC * C1C);
  fill_kernel<<<gridOf(N_NODESC), 256, 0, stream>>>(m2, -1e30f, N_NODESC);
  fill_kernel<<<gridOf(N_NODESC), 256, 0, stream>>>(den2, 0.0f, N_NODESC);
  fill_kernel<<<gridOf((long long)N_NODESC * NAPC), 256, 0, stream>>>(agg2, 0.0f, N_NODESC * NAPC);

  // self-loop edge attributes (segment mean by dst)
  loop_accum_kernel<<<gridOf(N_EDGESC), 256, 0, stream>>>(ei, eattr, cnt, asum);
  loop_fin_kernel<<<gridOf(N_NODESC), 256, 0, stream>>>(cnt, asum, lattr);

  // bf16 staging for WMMA operands
  cvt_bf16_kernel<<<gridOf(N_NODESC * F_INC), 256, 0, stream>>>(x, xbf, N_NODESC * F_INC, N_NODESC * F_INC);
  cvt_bf16_kernel<<<gridOf(C1C * F_INC), 256, 0, stream>>>(Wl1, w1l, C1C * F_INC, C1C * F_INC);
  cvt_bf16_kernel<<<gridOf(C1C * F_INC), 256, 0, stream>>>(Wr1, w1r, C1C * F_INC, C1C * F_INC);
  cvt_bf16_kernel<<<gridOf((long long)NAPC * C1C), 256, 0, stream>>>(Wl2, w2l, NAC * C1C, NAPC * C1C);
  cvt_bf16_kernel<<<gridOf((long long)NAPC * C1C), 256, 0, stream>>>(Wr2, w2r, NAC * C1C, NAPC * C1C);

  // layer-1 GEMMs: xl1/xr1 = x @ W^T + b   (125 M-tiles x 4 N64-groups = 500 waves)
  {
    int tiles = (N_NODESC / 16) * (C1C / 64);
    dim3 g((tiles + 3) / 4);
    wmma_gemm_kernel<F_INC, C1C, C1C><<<g, 128, 0, stream>>>(xbf, w1l, bl1, xl1, N_NODESC);
    wmma_gemm_kernel<F_INC, C1C, C1C><<<g, 128, 0, stream>>>(xbf, w1r, br1, xr1, N_NODESC);
  }

  // layer-1 attention + aggregation
  logits1_kernel<<<gridOf((long long)E2C * 4), 256, 0, stream>>>(ei, eattr, lattr, xl1, xr1, We1, att1, log1, m1);
  exp1_kernel<<<gridOf((long long)E2C * 4), 256, 0, stream>>>(ei, eattr, lattr, log1, m1, den1);
  scatter1_kernel<<<dim3(E2C), 256, 0, stream>>>(ei, eattr, lattr, log1, den1, xl1, agg1);
  h1_fin_kernel<<<gridOf((long long)N_NODESC * C1C), 256, 0, stream>>>(agg1, b1, h1bf);

  // layer-2 GEMMs: xl2/xr2 = h @ W^T + b   (125 x 94 = 11750 waves)
  {
    int tiles = (N_NODESC / 16) * (NAPC / 64);
    dim3 g((tiles + 3) / 4);
    wmma_gemm_kernel<C1C, NAPC, NAC><<<g, 128, 0, stream>>>(h1bf, w2l, bl2, xl2, N_NODESC);
    wmma_gemm_kernel<C1C, NAPC, NAC><<<g, 128, 0, stream>>>(h1bf, w2r, br2, xr2, N_NODESC);
  }

  // layer-2 attention + aggregation
  logits2_kernel<<<dim3(E2C), 256, 0, stream>>>(ei, eattr, lattr, xl2, xr2, We2, att2, log2, m2);
  exp2_kernel<<<gridOf(E2C), 256, 0, stream>>>(ei, eattr, lattr, log2, m2, den2);
  scatter2_kernel<<<dim3(E2C, (NAC + 255) / 256), 256, 0, stream>>>(ei, eattr, lattr, log2, den2, xl2, agg2);

  // pool + softmax + sigmoid(alpha)
  pool_kernel<<<gridOf(NAC), 256, 0, stream>>>(agg2, b2, pooled);
  softmax_out_kernel<<<1, 1024, 0, stream>>>(pooled, alpha, out);
}
